// DiffusionModel_80298708566729
// MI455X (gfx1250) — matrix operand order repaired and checked by reading, not compiled, not run on hardware
//
#include <hip/hip_runtime.h>
#include <hip/hip_bf16.h>
#include <cstdint>

typedef __attribute__((ext_vector_type(16))) __bf16 v16bf;
typedef __attribute__((ext_vector_type(8)))  float  v8f;

#define HW   64
#define NB   8
#define NPIX (NB * HW * HW)      /* 32768 pixels total */
#define SIGMA 0.796921f          /* sqrt(sum_{t=2..101} 1/t^2) */

#if defined(__has_builtin)
# if __has_builtin(__builtin_amdgcn_global_load_async_to_lds_b128)
#  define HAVE_ASYNC_COPY 1
# endif
#endif
#ifndef HAVE_ASYNC_COPY
# define HAVE_ASYNC_COPY 0
#endif

#if HAVE_ASYNC_COPY
/* param 1 of the builtin is 'int __vector_size__(16) __device__ *'
 * (addrspace(1) v4i*); LDS side symmetric in addrspace(3). */
typedef int v4i_async __attribute__((vector_size(16)));
typedef __attribute__((address_space(1))) v4i_async* async_gptr;
typedef __attribute__((address_space(3))) v4i_async* async_lptr;
#endif

/* ---------------- small helpers ---------------- */

__device__ __forceinline__ unsigned pcg(unsigned v) {
    v = v * 747796405u + 2891336453u;
    unsigned w = ((v >> ((v >> 28u) + 4u)) ^ v) * 277803737u;
    return (w >> 22u) ^ w;
}

__device__ __forceinline__ float gaussian(unsigned id) {
    unsigned h1 = pcg(id * 2u + 1u), h2 = pcg(id * 2u + 2u);
    float u1 = ((float)h1 + 1.0f) * 2.3283064e-10f;   /* (0,1] */
    float u2 = (float)h2 * 2.3283064e-10f;
    return sqrtf(-2.0f * __logf(u1)) * __cosf(6.2831853f * u2);
}

/* lp[b*4096+hw] = label[b]*fc_w[hw] + fc_b[hw] */
__global__ void lp_kernel(const float* __restrict__ label,
                          const float* __restrict__ fcw,
                          const float* __restrict__ fcb,
                          float* __restrict__ lp) {
    int idx = blockIdx.x * 256 + threadIdx.x;
    if (idx >= NPIX) return;
    int b = idx >> 12, hw = idx & 4095;
    lp[idx] = label[b] * fcw[hw] + fcb[hw];
}

/* inp NHWC bf16, 288 channels: [0..255]=x (emb + 100*lp + noise), [256]=lp, [257..287]=0 */
__global__ void build_inp(const float* __restrict__ emb,
                          const float* __restrict__ lp,
                          __bf16* __restrict__ inp) {
    int idx = blockIdx.x * 256 + threadIdx.x;
    if (idx >= NPIX * 288) return;
    int p = idx / 288, c = idx - p * 288;
    int b = p >> 12, hw = p & 4095;
    float v;
    float lpv = lp[p];
    if (c < 256) {
        float e = emb[(b * 256 + c) * 4096 + hw];
        v = e + 100.0f * lpv + SIGMA * gaussian((unsigned)(p * 256 + c));
    } else if (c == 256) {
        v = lpv;
    } else {
        v = 0.0f;
    }
    inp[idx] = (__bf16)v;
}

/* OIHW f32 -> [tap][O][Ipad] bf16 (Ipad >= Ireal, zero pad) */
__global__ void repack_w(const float* __restrict__ src, __bf16* __restrict__ dst,
                         int O, int Ireal, int Ipad, int ksize) {
    int idx = blockIdx.x * 256 + threadIdx.x;
    int total = ksize * ksize * O * Ipad;
    if (idx >= total) return;
    int tap = idx / (O * Ipad);
    int r = idx - tap * (O * Ipad);
    int o = r / Ipad;
    int i = r - o * Ipad;
    int ky = tap / ksize, kx = tap - ky * ksize;
    float v = (i < Ireal) ? src[((o * Ireal + i) * ksize + ky) * ksize + kx] : 0.0f;
    dst[idx] = (__bf16)v;
}

/* mlab NHWC bf16, 160 ch: [0..127]=mid, [128]=label[b], [129..159]=0 */
__global__ void build_mlab(const __bf16* __restrict__ mid,
                           const float* __restrict__ label,
                           __bf16* __restrict__ mlab) {
    int idx = blockIdx.x * 256 + threadIdx.x;
    if (idx >= NPIX * 160) return;
    int p = idx / 160, c = idx - p * 160;
    __bf16 v;
    if (c < 128)       v = mid[p * 128 + c];
    else if (c == 128) v = (__bf16)label[p >> 12];
    else               v = (__bf16)0.0f;
    mlab[idx] = v;
}

/* pooled[b*160+c] = mean over 4096 pixels of mlab channel c */
__global__ void pool_mlab(const __bf16* __restrict__ mlab, float* __restrict__ pooled) {
    int b = blockIdx.x / 160, c = blockIdx.x % 160;
    __shared__ float sm[256];
    float s = 0.0f;
    for (int i = threadIdx.x; i < 4096; i += 256)
        s += (float)mlab[(b * 4096 + i) * 160 + c];
    sm[threadIdx.x] = s;
    __syncthreads();
    for (int st = 128; st > 0; st >>= 1) {
        if ((int)threadIdx.x < st) sm[threadIdx.x] += sm[threadIdx.x + st];
        __syncthreads();
    }
    if (threadIdx.x == 0) pooled[b * 160 + c] = sm[0] * (1.0f / 4096.0f);
}

/* atts[b*128+o] = sigmoid(dot(pooled[b,0:129], satt_w[o]) + satt_b[o]) */
__global__ void satt_kernel(const float* __restrict__ pooled,
                            const float* __restrict__ w,
                            const float* __restrict__ bias,
                            float* __restrict__ atts) {
    int t = blockIdx.x * 256 + threadIdx.x;
    if (t >= 1024) return;
    int b = t >> 7, o = t & 127;
    float s = bias[o];
    for (int i = 0; i < 129; ++i) s += pooled[b * 160 + i] * w[o * 129 + i];
    atts[t] = 1.0f / (1.0f + __expf(-s));
}

/* in-place: a[i] (= att_d) *= mid[i]  -> md */
__global__ void mul_dense(const __bf16* __restrict__ mid, __bf16* __restrict__ a) {
    int i = blockIdx.x * 256 + threadIdx.x;
    if (i >= NPIX * 128) return;
    a[i] = (__bf16)((float)mid[i] * (float)a[i]);
}

/* ms[i] = mid[i] * atts[b,c] */
__global__ void mul_sparse(const __bf16* __restrict__ mid,
                           const float* __restrict__ atts,
                           __bf16* __restrict__ ms) {
    int i = blockIdx.x * 256 + threadIdx.x;
    if (i >= NPIX * 128) return;
    int c = i & 127;
    int b = i >> 19;
    ms[i] = (__bf16)((float)mid[i] * atts[b * 128 + c]);
}

/* sparse out[b,g,c] = mean over (32 rows, 64 cols) of se channel c -> d_out tail */
__global__ void sparse_pool(const __bf16* __restrict__ se, float* __restrict__ out) {
    int c = blockIdx.x & 255;
    int bg = blockIdx.x >> 8;
    int b = bg >> 1, g = bg & 1;
    __shared__ float sm[256];
    float s = 0.0f;
    for (int j = threadIdx.x; j < 2048; j += 256) {
        int y = g * 32 + (j >> 6), x = j & 63;
        s += (float)se[((b * HW + y) * HW + x) * 256 + c];
    }
    sm[threadIdx.x] = s;
    __syncthreads();
    for (int st = 128; st > 0; st >>= 1) {
        if ((int)threadIdx.x < st) sm[threadIdx.x] += sm[threadIdx.x + st];
        __syncthreads();
    }
    if (threadIdx.x == 0) out[(b * 2 + g) * 256 + c] = sm[0] * (1.0f / 2048.0f);
}

/* 16-byte copy global -> LDS: async (ASYNCcnt) when available, else sync */
__device__ __forceinline__ void copy_b128_to_lds(const __bf16* gsrc, __bf16* ldst) {
#if HAVE_ASYNC_COPY
    __builtin_amdgcn_global_load_async_to_lds_b128(
        (async_gptr)(uintptr_t)gsrc,
        (async_lptr)(uint32_t)(uintptr_t)ldst,
        0, 0);
#else
    *(uint4*)ldst = *(const uint4*)gsrc;
#endif
}

__device__ __forceinline__ void wait_async_copies() {
#if HAVE_ASYNC_COPY
# if __has_builtin(__builtin_amdgcn_s_wait_asynccnt)
    __builtin_amdgcn_s_wait_asynccnt(0);
# else
    asm volatile("s_wait_asynccnt 0" ::: "memory");
# endif
#endif
}

/* ---------------- main WMMA conv (implicit GEMM) ----------------
 * Activations NHWC bf16. Block = 8 waves: 128 consecutive pixels (2 rows)
 * x one 64-outchannel group. Per (tap, 32-K chunk) the 64x32 bf16 weight
 * tile (4KB) is staged into LDS once per block (async -> ASYNCcnt when
 * available), double buffered, one barrier per chunk. Each wave runs 4
 * V_WMMA_F32_16X16X32_BF16 per chunk with the A fragment reused 4x.
 * Out-of-image rows/cols contribute zero A fragments (keeps barriers
 * block-uniform). wt layout: [taps][cout][cin] bf16, cin % 32 == 0.
 */
__global__ __launch_bounds__(256) void conv_wmma(
    const __bf16* __restrict__ in0, int c0,
    const __bf16* __restrict__ in1, int cin,
    const __bf16* __restrict__ wt, const float* __restrict__ bias,
    int cout, int ksize, int relu, int f32nchw, int ngshift,
    __bf16* __restrict__ outb, float* __restrict__ outf) {
    __shared__ __bf16 sB[2][64 * 32];              /* 8 KB double buffer */

    const int lane = threadIdx.x & 31;
    const int wv   = threadIdx.x >> 5;
    const int ngrp   = blockIdx.x & ((1 << ngshift) - 1);
    const int mblock = blockIdx.x >> ngshift;      /* 256 pixel-blocks of 128 px */
    const int b  = mblock >> 5;
    const int y  = ((mblock & 31) << 1) + (wv >> 2);
    const int x0 = (wv & 3) << 4;
    const int hi = lane >> 4;
    const int lo = lane & 15;
    const int x  = x0 + lo;
    const int nbase0 = ngrp << 6;

    /* staging role of this thread: one b128 per chunk */
    const int sn = threadIdx.x >> 2;               /* row 0..63 of weight tile */
    const int sk = (threadIdx.x & 3) << 3;         /* k offset 0,8,16,24 */

    v8f acc0 = {0.f,0.f,0.f,0.f,0.f,0.f,0.f,0.f};
    v8f acc1 = acc0, acc2 = acc0, acc3 = acc0;
    const int taps = ksize * ksize;
    const int c1   = cin - c0;

    int buf = 0;
    for (int tap = 0; tap < taps; ++tap) {
        const int dy = (ksize == 3) ? (tap / 3 - 1) : 0;
        const int dx = (ksize == 3) ? (tap % 3 - 1) : 0;
        const int ys = y + dy;
        const int xs = x + dx;
        const bool inb = (ys >= 0) && (ys < HW) && (xs >= 0) && (xs < HW);
        const int ysc = (ys < 0) ? 0 : (ys > HW - 1 ? HW - 1 : ys);
        const int xsc = (xs < 0) ? 0 : (xs > HW - 1 ? HW - 1 : xs);
        const int prow = (b * HW + ysc) * HW + xsc;
        const __bf16* wtap = wt + (tap * cout + nbase0) * cin;

        for (int kc = 0; kc < cin; kc += 32) {
            /* ---- stage B chunk [64 x 32] into LDS (1 b128 per thread) ---- */
            copy_b128_to_lds(wtap + sn * cin + kc + sk, &sB[buf][sn * 32 + sk]);
            wait_async_copies();
            __syncthreads();

            /* ---- A fragment: direct global (L2-resident activations) ---- */
            const __bf16* src; int cl, cs;
            if (kc < c0) { src = in0; cl = kc;      cs = c0; }
            else         { src = in1; cl = kc - c0; cs = c1; }
            union { v16bf v; uint4 q[2]; } A;
            if (inb) {
                const __bf16* ap = src + prow * cs + cl + hi * 8;
                A.q[0] = *(const uint4*)(ap);
                A.q[1] = *(const uint4*)(ap + 16);
            } else {
                uint4 z = {0u, 0u, 0u, 0u};
                A.q[0] = z; A.q[1] = z;
            }

            union { v16bf v; uint4 q[2]; } Bm;
#define DO_WMMA(S, ACC)                                                        \
            {                                                                  \
                const uint4* bq =                                              \
                    (const uint4*)&sB[buf][((S) * 16 + lo) * 32 + hi * 8];     \
                Bm.q[0] = bq[0];                                               \
                Bm.q[1] = bq[2];                                               \
                ACC = __builtin_amdgcn_wmma_f32_16x16x32_bf16(                 \
                    false, A.v, false, Bm.v, (short)0, ACC, false, false);     \
            }
            DO_WMMA(0, acc0)
            DO_WMMA(1, acc1)
            DO_WMMA(2, acc2)
            DO_WMMA(3, acc3)
#undef DO_WMMA
            buf ^= 1;
        }
    }

    /* epilogue: bias + optional ReLU; D layout: lane holds N=lo, rows M=v+8*hi */
    v8f accs[4] = {acc0, acc1, acc2, acc3};
#pragma unroll
    for (int s = 0; s < 4; ++s) {
        const int nglob = nbase0 + s * 16 + lo;
        const float bv = bias[nglob];
#pragma unroll
        for (int v = 0; v < 8; ++v) {
            float r = accs[s][v] + bv;
            if (relu) r = fmaxf(r, 0.0f);
            const int xo = x0 + v + 8 * hi;
            if (f32nchw) {
                outf[(b * cout + nglob) * (HW * HW) + y * HW + xo] = r;
            } else {
                outb[((b * HW + y) * HW + xo) * cout + nglob] = (__bf16)r;
            }
        }
    }
}

/* ---------------- host orchestration ---------------- */

extern "C" void kernel_launch(void* const* d_in, const int* in_sizes, int n_in,
                              void* d_out, int out_size, void* d_ws, size_t ws_size,
                              hipStream_t stream) {
    (void)in_sizes; (void)n_in; (void)out_size; (void)ws_size;

    const float* emb   = (const float*)d_in[0];
    const float* label = (const float*)d_in[1];
    const float* fcw   = (const float*)d_in[2];
    const float* fcb   = (const float*)d_in[3];
    const float* enc1_w1 = (const float*)d_in[4];  const float* enc1_b1 = (const float*)d_in[5];
    const float* enc1_w2 = (const float*)d_in[6];  const float* enc1_b2 = (const float*)d_in[7];
    const float* enc2_w1 = (const float*)d_in[8];  const float* enc2_b1 = (const float*)d_in[9];
    const float* enc2_w2 = (const float*)d_in[10]; const float* enc2_b2 = (const float*)d_in[11];
    const float* mid_w1  = (const float*)d_in[12]; const float* mid_b1  = (const float*)d_in[13];
    const float* mid_w2  = (const float*)d_in[14]; const float* mid_b2  = (const float*)d_in[15];
    const float* datt_w  = (const float*)d_in[16]; const float* datt_b  = (const float*)d_in[17];
    const float* ddec1_w1 = (const float*)d_in[18]; const float* ddec1_b1 = (const float*)d_in[19];
    const float* ddec1_w2 = (const float*)d_in[20]; const float* ddec1_b2 = (const float*)d_in[21];
    const float* ddec2_w1 = (const float*)d_in[22]; const float* ddec2_b1 = (const float*)d_in[23];
    const float* ddec2_w2 = (const float*)d_in[24]; const float* ddec2_b2 = (const float*)d_in[25];
    const float* satt_w   = (const float*)d_in[26]; const float* satt_b   = (const float*)d_in[27];
    const float* sdec1_w1 = (const float*)d_in[28]; const float* sdec1_b1 = (const float*)d_in[29];
    const float* sdec1_w2 = (const float*)d_in[30]; const float* sdec1_b2 = (const float*)d_in[31];
    const float* sdec2_w1 = (const float*)d_in[32]; const float* sdec2_b1 = (const float*)d_in[33];
    const float* sdec2_w2 = (const float*)d_in[34]; const float* sdec2_b2 = (const float*)d_in[35];

    float* outf = (float*)d_out;               /* dense [8,256,64,64], then sparse [8,2,256] */
    float* out_sparse = outf + 8 * 256 * 4096;

    size_t off = 0;
    char* base = (char*)d_ws;
    auto take = [&](size_t bytes) -> char* {
        char* p = base + off;
        off += (bytes + 255) & ~(size_t)255;
        return p;
    };
    float*  lp     = (float*)take((size_t)NPIX * 4);
    float*  pooled = (float*)take(8 * 160 * 4);
    float*  atts   = (float*)take(8 * 128 * 4);

    __bf16* wEnc1a  = (__bf16*)take((size_t)9 * 64 * 288 * 2);
    __bf16* wEnc1b  = (__bf16*)take((size_t)9 * 64 * 64 * 2);
    __bf16* wEnc2a  = (__bf16*)take((size_t)9 * 128 * 64 * 2);
    __bf16* wEnc2b  = (__bf16*)take((size_t)9 * 128 * 128 * 2);
    __bf16* wMida   = (__bf16*)take((size_t)9 * 256 * 128 * 2);
    __bf16* wMidb   = (__bf16*)take((size_t)9 * 128 * 256 * 2);
    __bf16* wDatt   = (__bf16*)take((size_t)1 * 128 * 160 * 2);
    __bf16* wDd1a   = (__bf16*)take((size_t)9 * 128 * 256 * 2);
    __bf16* wDd1b   = (__bf16*)take((size_t)9 * 64 * 128 * 2);
    __bf16* wDd2a   = (__bf16*)take((size_t)9 * 64 * 128 * 2);
    __bf16* wDd2b   = (__bf16*)take((size_t)9 * 256 * 64 * 2);
    __bf16* wSd1a   = (__bf16*)take((size_t)9 * 128 * 256 * 2);
    __bf16* wSd1b   = (__bf16*)take((size_t)9 * 64 * 128 * 2);
    __bf16* wSd2a   = (__bf16*)take((size_t)9 * 64 * 128 * 2);
    __bf16* wSd2b   = (__bf16*)take((size_t)9 * 256 * 64 * 2);

    __bf16* INP  = (__bf16*)take((size_t)NPIX * 288 * 2);  /* reused as MLAB, then SE */
    __bf16* E1   = (__bf16*)take((size_t)NPIX * 64 * 2);
    __bf16* E2   = (__bf16*)take((size_t)NPIX * 128 * 2);
    __bf16* MID  = (__bf16*)take((size_t)NPIX * 128 * 2);
    __bf16* T1   = (__bf16*)take((size_t)NPIX * 256 * 2);
    __bf16* ABUF = (__bf16*)take((size_t)NPIX * 128 * 2);  /* att_d -> md -> ms */
    __bf16* BBUF = (__bf16*)take((size_t)NPIX * 64 * 2);   /* d1 / s1 */
    __bf16* MLAB = INP;
    __bf16* SE   = INP;

    auto repack = [&](const float* src, __bf16* dst, int O, int Ir, int Ip, int ks) {
        int total = ks * ks * O * Ip;
        repack_w<<<(total + 255) / 256, 256, 0, stream>>>(src, dst, O, Ir, Ip, ks);
    };
    auto conv = [&](const __bf16* i0, int c0, const __bf16* i1, int cin,
                    const __bf16* w, const float* bs, int cout, int ks, int relu,
                    __bf16* ob, float* of) {
        int ngshift = (cout == 64) ? 0 : (cout == 128) ? 1 : 2;
        int blocks = 256 << ngshift;               /* 256 pixel-blocks x cout/64 */
        conv_wmma<<<blocks, 256, 0, stream>>>(i0, c0, i1, cin, w, bs,
                                              cout, ks, relu, of ? 1 : 0, ngshift,
                                              ob, of);
    };

    /* weight repack (f32 OIHW -> bf16 [tap][O][Ipad]) */
    repack(enc1_w1, wEnc1a, 64, 257, 288, 3);
    repack(enc1_w2, wEnc1b, 64, 64, 64, 3);
    repack(enc2_w1, wEnc2a, 128, 64, 64, 3);
    repack(enc2_w2, wEnc2b, 128, 128, 128, 3);
    repack(mid_w1,  wMida, 256, 128, 128, 3);
    repack(mid_w2,  wMidb, 128, 256, 256, 3);
    repack(datt_w,  wDatt, 128, 129, 160, 1);
    repack(ddec1_w1, wDd1a, 128, 256, 256, 3);
    repack(ddec1_w2, wDd1b, 64, 128, 128, 3);
    repack(ddec2_w1, wDd2a, 64, 128, 128, 3);
    repack(ddec2_w2, wDd2b, 256, 64, 64, 3);
    repack(sdec1_w1, wSd1a, 128, 256, 256, 3);
    repack(sdec1_w2, wSd1b, 64, 128, 128, 3);
    repack(sdec2_w1, wSd2a, 64, 128, 128, 3);
    repack(sdec2_w2, wSd2b, 256, 64, 64, 3);

    /* diffusion prefix + input assembly */
    lp_kernel<<<(NPIX + 255) / 256, 256, 0, stream>>>(label, fcw, fcb, lp);
    build_inp<<<((size_t)NPIX * 288 + 255) / 256, 256, 0, stream>>>(emb, lp, INP);

    /* encoder */
    conv(INP, 288, nullptr, 288, wEnc1a, enc1_b1, 64, 3, 1, T1, nullptr);
    conv(T1, 64, nullptr, 64, wEnc1b, enc1_b2, 64, 3, 1, E1, nullptr);
    conv(E1, 64, nullptr, 64, wEnc2a, enc2_b1, 128, 3, 1, T1, nullptr);
    conv(T1, 128, nullptr, 128, wEnc2b, enc2_b2, 128, 3, 1, E2, nullptr);
    conv(E2, 128, nullptr, 128, wMida, mid_b1, 256, 3, 1, T1, nullptr);
    conv(T1, 256, nullptr, 256, wMidb, mid_b2, 128, 3, 1, MID, nullptr);

    /* attention inputs */
    build_mlab<<<((size_t)NPIX * 160 + 255) / 256, 256, 0, stream>>>(MID, label, MLAB);
    conv(MLAB, 160, nullptr, 160, wDatt, datt_b, 128, 1, 0, ABUF, nullptr); /* att_d */
    pool_mlab<<<8 * 160, 256, 0, stream>>>(MLAB, pooled);
    satt_kernel<<<4, 256, 0, stream>>>(pooled, satt_w, satt_b, atts);

    /* dense branch */
    mul_dense<<<((size_t)NPIX * 128 + 255) / 256, 256, 0, stream>>>(MID, ABUF); /* md */
    conv(ABUF, 128, E2, 256, wDd1a, ddec1_b1, 128, 3, 1, T1, nullptr);
    conv(T1, 128, nullptr, 128, wDd1b, ddec1_b2, 64, 3, 1, BBUF, nullptr);      /* d1 */
    conv(BBUF, 64, E1, 128, wDd2a, ddec2_b1, 64, 3, 1, T1, nullptr);
    conv(T1, 64, nullptr, 64, wDd2b, ddec2_b2, 256, 3, 1, nullptr, outf);       /* dense */

    /* sparse branch */
    mul_sparse<<<((size_t)NPIX * 128 + 255) / 256, 256, 0, stream>>>(MID, atts, ABUF);
    conv(ABUF, 128, E2, 256, wSd1a, sdec1_b1, 128, 3, 1, T1, nullptr);
    conv(T1, 128, nullptr, 128, wSd1b, sdec1_b2, 64, 3, 1, BBUF, nullptr);      /* s1 */
    conv(BBUF, 64, E1, 128, wSd2a, sdec2_b1, 64, 3, 1, T1, nullptr);
    conv(T1, 64, nullptr, 64, wSd2b, sdec2_b2, 256, 3, 1, SE, nullptr);         /* se */

    sparse_pool<<<8 * 2 * 256, 256, 0, stream>>>(SE, out_sparse);
}
